// ZambaMambaMixer_2473901163085
// MI455X (gfx1250) — matrix operand order repaired and checked by reading, not compiled, not run on hardware
//
#include <hip/hip_runtime.h>
#include <hip/hip_bf16.h>
#include <stdint.h>

#define USE_ASYNC 1

// ---------------------------------------------------------------- dimensions
#define BATCH   2
#define SEQ     2048
#define HDIM    2048
#define IDIM    4096      // EXPAND*H
#define ODIM    8192      // 2*I
#define NHEADS  2
#define HEADD   2048      // HD
#define NSTATE  16
#define RRANK   128
#define KCONV   4

// ---------------------------------------------------------------- WMMA types
typedef __attribute__((ext_vector_type(16))) __bf16 v16bf;
typedef __attribute__((ext_vector_type(8)))  float  v8f;

// GEMM tiling: block tile 128x128, 8 waves (wave32), each wave 64(M)x32(N)
#define BM 128
#define BN 128
#define BK 32
#define LDSK 40   // padded K stride (bf16 elems) to spread LDS banks

// branch-free RNE f32 -> bf16
__device__ __forceinline__ uint32_t bf16_1(float a) {
    uint32_t u = __builtin_bit_cast(uint32_t, a);
    return (u + 0x7FFFu + ((u >> 16) & 1u)) >> 16;
}
__device__ __forceinline__ uint32_t pk_bf16(float a, float b) {
    uint32_t ua = __builtin_bit_cast(uint32_t, a);
    uint32_t ub = __builtin_bit_cast(uint32_t, b);
    ua = (ua + 0x7FFFu + ((ua >> 16) & 1u)) >> 16;
    ub = (ub + 0x7FFFu + ((ub >> 16) & 1u)) & 0xFFFF0000u;
    return ua | ub;
}

union F4 { float4 v; float f[4]; };

// ------------------------------------------------------------------ GEMM
// C[m,n] = sum_k A[m,k]*B[k,n]; sa_k==1 required; B contiguous along n (bnc=1)
// or along k (bnc=0). epilogue 1: softplus(acc + bias[zo*bias_zo + m])
__global__ __launch_bounds__(256, 2)
void gemm_bf16_wmma(const float* __restrict__ A,  long sa_m, long a_zo, long a_zi,
                    const float* __restrict__ Bg, long sb_k, long sb_n, long b_zo, long b_zi,
                    float* __restrict__ C,        long sc_m, long sc_n, long c_zo, long c_zi,
                    int M, int N, int Kd, int innerCnt, int bnc,
                    int epilogue, const float* __restrict__ bias, long bias_zo)
{
    __shared__ unsigned short As[BM * LDSK];
    __shared__ unsigned short Bs[BN * LDSK];

    const int tid  = threadIdx.x;
    const int z    = blockIdx.z;
    const int zo   = z / innerCnt;
    const int zi   = z % innerCnt;
    const float* Ab = A  + (long)zo * a_zo + (long)zi * a_zi;
    const float* Bb = Bg + (long)zo * b_zo + (long)zi * b_zi;
    float*       Cb = C  + (long)zo * c_zo + (long)zi * c_zi;

    const int m0 = blockIdx.y * BM;
    const int n0 = blockIdx.x * BN;

    const int w    = tid >> 5;
    const int lane = tid & 31;
    const int wm   = (w >> 2) * 64;   // wave M origin (0/64)
    const int wn   = (w & 3)  * 32;   // wave N origin (0/32/64/96)
    const int lg   = lane >> 4;
    const int l16  = lane & 15;

    v8f acc[4][2];
    const v8f vzero = {0.f,0.f,0.f,0.f,0.f,0.f,0.f,0.f};
    #pragma unroll
    for (int i = 0; i < 4; i++)
        #pragma unroll
        for (int j = 0; j < 2; j++) acc[i][j] = vzero;

    F4 arg[4], brg[4];

    auto preload = [&](int k0) {
        #pragma unroll
        for (int j = 0; j < 4; j++) {            // A: 1024 float4, k fast
            int f = j * 256 + tid;
            int m = f >> 3, kq = f & 7;
            if (m0 + m < M)
                arg[j].v = *(const float4*)(Ab + (long)(m0 + m) * sa_m + k0 + kq * 4);
            else { arg[j].f[0]=0.f; arg[j].f[1]=0.f; arg[j].f[2]=0.f; arg[j].f[3]=0.f; }
        }
        if (k0 + 2 * BK < Kd)                    // global_prefetch of tile k+2
            __builtin_prefetch(Ab + (long)(m0 + (tid >> 1)) * sa_m + k0 + 2 * BK, 0, 1);
        if (bnc) {
            #pragma unroll
            for (int j = 0; j < 4; j++) {        // n contiguous: 32k x 128n
                int f = j * 256 + tid;
                int k = f >> 5, nq = f & 31;
                brg[j].v = *(const float4*)(Bb + (long)(k0 + k) * sb_k + n0 + nq * 4);
            }
        } else {
            #pragma unroll
            for (int j = 0; j < 4; j++) {        // k contiguous: 128n x 32k
                int f = j * 256 + tid;
                int n = f >> 3, kq = f & 7;
                brg[j].v = *(const float4*)(Bb + (long)(n0 + n) * sb_n + k0 + kq * 4);
            }
        }
    };

    auto stage = [&]() {
        #pragma unroll
        for (int j = 0; j < 4; j++) {
            int f = j * 256 + tid;
            int m = f >> 3, kq = f & 7;
            uint32_t* dst = (uint32_t*)&As[m * LDSK + kq * 4];
            dst[0] = pk_bf16(arg[j].f[0], arg[j].f[1]);
            dst[1] = pk_bf16(arg[j].f[2], arg[j].f[3]);
        }
        if (bnc) {
            #pragma unroll
            for (int j = 0; j < 4; j++) {
                int f = j * 256 + tid;
                int k = f >> 5, nq = f & 31;
                #pragma unroll
                for (int i = 0; i < 4; i++)
                    Bs[(nq * 4 + i) * LDSK + k] = (unsigned short)bf16_1(brg[j].f[i]);
            }
        } else {
            #pragma unroll
            for (int j = 0; j < 4; j++) {
                int f = j * 256 + tid;
                int n = f >> 3, kq = f & 7;
                uint32_t* dst = (uint32_t*)&Bs[n * LDSK + kq * 4];
                dst[0] = pk_bf16(brg[j].f[0], brg[j].f[1]);
                dst[1] = pk_bf16(brg[j].f[2], brg[j].f[3]);
            }
        }
    };

    preload(0);
    for (int k0 = 0; k0 < Kd; k0 += BK) {
        if (k0) __syncthreads();                 // LDS free after previous wmma reads
        stage();
        __syncthreads();
        if (k0 + BK < Kd) preload(k0 + BK);      // overlap next-tile loads with wmma

        union Fr { v16bf v; uint32_t u[8]; };
        Fr afr[4];
        #pragma unroll
        for (int tm = 0; tm < 4; tm++) {
            int m = wm + tm * 16 + l16;
            const uint32_t* row = (const uint32_t*)&As[m * LDSK];
            #pragma unroll
            for (int vv = 0; vv < 8; vv++) {
                int kk = ((vv < 4) ? 2 * vv : 16 + 2 * (vv - 4)) + lg * 8;
                afr[tm].u[vv] = row[kk >> 1];
            }
        }
        #pragma unroll
        for (int tn = 0; tn < 2; tn++) {
            Fr bfr;
            int n = wn + tn * 16 + l16;
            const uint32_t* row = (const uint32_t*)&Bs[n * LDSK];
            #pragma unroll
            for (int vv = 0; vv < 8; vv++) {
                int kk = ((vv < 4) ? 2 * vv : 16 + 2 * (vv - 4)) + lg * 8;
                bfr.u[vv] = row[kk >> 1];
            }
            #pragma unroll
            for (int tm = 0; tm < 4; tm++) {
                acc[tm][tn] = __builtin_amdgcn_wmma_f32_16x16x32_bf16(
                    false, afr[tm].v, false, bfr.v, (short)0, acc[tm][tn], false, false);
            }
        }
    }

    #pragma unroll
    for (int tm = 0; tm < 4; tm++)
        #pragma unroll
        for (int tn = 0; tn < 2; tn++)
            #pragma unroll
            for (int r = 0; r < 8; r++) {
                int m = m0 + wm + tm * 16 + r + 8 * lg;
                int n = n0 + wn + tn * 16 + l16;
                if (m < M) {
                    float v = acc[tm][tn][r];
                    if (epilogue == 1) {
                        v += bias[(long)zo * bias_zo + m];
                        v = (v > 20.f) ? v : log1pf(__expf(v));
                    }
                    Cb[(long)m * sc_m + (long)n * sc_n] = v;
                }
            }
}

// ---------------------------------------------- depthwise causal conv + SiLU
__global__ void conv_silu_kernel(const float* __restrict__ proj,
                                 const float* __restrict__ cw,
                                 const float* __restrict__ cb,
                                 float* __restrict__ xc)
{
    int s = blockIdx.x * 256 + threadIdx.x;
    int i = blockIdx.y;
    int b = blockIdx.z;
    const float* xin = proj + ((long)b * ODIM + 2 * (long)i) * SEQ;
    float acc = cb[i];
    #pragma unroll
    for (int k = 0; k < KCONV; k++) {
        int sp = s - (KCONV - 1) + k;
        if (sp >= 0) acc += xin[sp] * cw[i * KCONV + k];
    }
    float sig = 1.f / (1.f + __expf(-acc));
    xc[((long)b * IDIM + i) * SEQ + s] = acc * sig;
}

// --------------------------------------------------------- selective scan
#define SCH 16

#if USE_ASYNC
__device__ __forceinline__ uint32_t lds_off(const void* p) {
    return (uint32_t)(uintptr_t)p;   // flat->LDS: addr[31:0] is the LDS offset
}
__device__ __forceinline__ void async_ld_b128(uint32_t ldso, const void* g) {
    asm volatile("global_load_async_to_lds_b128 %0, %1, off"
                 :: "v"(ldso), "v"((uint64_t)(uintptr_t)g) : "memory");
}
__device__ __forceinline__ void async_st_b128(uint32_t ldso, void* g) {
    asm volatile("global_store_async_from_lds_b128 %0, %1, off"
                 :: "v"((uint64_t)(uintptr_t)g), "v"(ldso) : "memory");
}
__device__ __forceinline__ void async_wait0() {
    asm volatile("s_wait_asynccnt 0x0" ::: "memory");
}
#endif

__global__ __launch_bounds__(256, 2)
void scan_kernel(const float* __restrict__ ssm_p,  // (h,b,160,S)
                 const float* __restrict__ dt,     // (h,b,d,S)
                 const float* __restrict__ xc,     // (b,i,S)
                 const float* __restrict__ A_log,  // (h,d,N)
                 float* __restrict__ y)            // (h,b,d,S)
{
    __shared__ float sB[NSTATE][SCH];   // [n][s]
    __shared__ float sC[NSTATE][SCH];
    __shared__ float sdt[256 * SCH];
    __shared__ float sx [256 * SCH];

    const int tid = threadIdx.x;
    const int d0  = blockIdx.x * 256;
    const int d   = d0 + tid;
    const int b   = blockIdx.y;
    const int h   = blockIdx.z;
    const long hb = (long)h * BATCH + b;

    const float* Bp  = ssm_p + (hb * 160 + RRANK) * (long)SEQ;
    const float* Cp  = ssm_p + (hb * 160 + RRANK + NSTATE) * (long)SEQ;
    const float* dtp = dt + hb * (long)HEADD * SEQ;
    const float* xp  = xc + ((long)b * IDIM + (long)h * HEADD) * SEQ;
    float*       yp  = y  + hb * (long)HEADD * SEQ;

    float a[NSTATE], st[NSTATE];
    #pragma unroll
    for (int n = 0; n < NSTATE; n++) {
        a[n]  = -__expf(A_log[((long)h * HEADD + d) * NSTATE + n]);
        st[n] = 0.f;
    }

    const int sq = (tid & 3) * 4;   // 16B column this thread copies
    const int dq = tid >> 2;        // row group

    for (int s0 = 0; s0 < SEQ; s0 += SCH) {
#if USE_ASYNC
        if (tid < 64) {             // B/C chunk: 16 rows x 16 steps
            int n = tid >> 2;
            async_ld_b128(lds_off(&sB[n][sq]), Bp + (long)n * SEQ + s0 + sq);
            async_ld_b128(lds_off(&sC[n][sq]), Cp + (long)n * SEQ + s0 + sq);
        }
        #pragma unroll
        for (int dd = 0; dd < 4; dd++) {
            int dr = dd * 64 + dq;
            async_ld_b128(lds_off(&sdt[dr * SCH + sq]), dtp + (long)(d0 + dr) * SEQ + s0 + sq);
            async_ld_b128(lds_off(&sx [dr * SCH + sq]), xp  + (long)(d0 + dr) * SEQ + s0 + sq);
        }
        async_wait0();
        __syncthreads();
#else
        if (tid < 64) {
            int n = tid >> 2;
            *(float4*)&sB[n][sq] = *(const float4*)(Bp + (long)n * SEQ + s0 + sq);
            *(float4*)&sC[n][sq] = *(const float4*)(Cp + (long)n * SEQ + s0 + sq);
        }
        #pragma unroll
        for (int dd = 0; dd < 4; dd++) {
            int dr = dd * 64 + dq;
            *(float4*)&sdt[dr * SCH + sq] = *(const float4*)(dtp + (long)(d0 + dr) * SEQ + s0 + sq);
            *(float4*)&sx [dr * SCH + sq] = *(const float4*)(xp  + (long)(d0 + dr) * SEQ + s0 + sq);
        }
        __syncthreads();
#endif
        float yv[SCH];
        #pragma unroll
        for (int ss = 0; ss < SCH; ss++) {
            float dtv = sdt[tid * SCH + ss];
            float xv  = sx [tid * SCH + ss];
            float accy = 0.f;
            #pragma unroll
            for (int n = 0; n < NSTATE; n++) {
                float dA = __expf(a[n] * dtv);
                st[n] = dA * st[n] + dtv * sB[n][ss] * xv;
                accy += st[n] * sC[n][ss];
            }
            yv[ss] = accy;
        }
        __syncthreads();
        #pragma unroll
        for (int ss = 0; ss < SCH; ss++) sdt[tid * SCH + ss] = yv[ss];
        __syncthreads();
#if USE_ASYNC
        #pragma unroll
        for (int dd = 0; dd < 4; dd++) {
            int dr = dd * 64 + dq;
            async_st_b128(lds_off(&sdt[dr * SCH + sq]), yp + (long)(d0 + dr) * SEQ + s0 + sq);
        }
        async_wait0();
        __syncthreads();
#else
        #pragma unroll
        for (int dd = 0; dd < 4; dd++) {
            int dr = dd * 64 + dq;
            *(float4*)(yp + (long)(d0 + dr) * SEQ + s0 + sq) = *(const float4*)&sdt[dr * SCH + sq];
        }
        __syncthreads();
#endif
    }
}

// ---------------------------------------------------------------- gating
__global__ void gate_kernel(const float* __restrict__ y,
                            const float* __restrict__ xc,
                            const float* __restrict__ proj,
                            const float* __restrict__ D,
                            float* __restrict__ yg)
{
    int s = blockIdx.x * 256 + threadIdx.x;
    int i = blockIdx.y;
    int b = blockIdx.z;
    int h = i / HEADD, dch = i % HEADD;
    long hb = (long)h * BATCH + b;
    float yv = y[(hb * HEADD + dch) * (long)SEQ + s];
    float xv = xc[((long)b * IDIM + i) * SEQ + s];
    float g  = proj[((long)b * ODIM + 2 * (long)i + 1) * SEQ + s];
    float r  = (yv + xv * D[h * HEADD + dch]) * (g / (1.f + __expf(-g)));
    yg[((long)b * IDIM + i) * SEQ + s] = r;
}

// ---------------------------------------------------------------- launcher
extern "C" void kernel_launch(void* const* d_in, const int* in_sizes, int n_in,
                              void* d_out, int out_size, void* d_ws, size_t ws_size,
                              hipStream_t stream)
{
    const float* hidden   = (const float*)d_in[0];
    const float* in_w     = (const float*)d_in[1];
    const float* conv_w   = (const float*)d_in[2];
    const float* conv_b   = (const float*)d_in[3];
    const float* xproj_w  = (const float*)d_in[4];
    const float* dtproj_w = (const float*)d_in[5];
    const float* dtproj_b = (const float*)d_in[6];
    const float* A_log    = (const float*)d_in[7];
    const float* Dp       = (const float*)d_in[8];
    const float* out_w    = (const float*)d_in[9];
    float* out = (float*)d_out;

    float* ws = (float*)d_ws;
    float* proj  = ws;                                        // (B, 8192, S)
    float* xconv = proj  + (long)BATCH * ODIM * SEQ;          // (B, I, S)
    float* ssm_p = xconv + (long)BATCH * IDIM * SEQ;          // (NH, B, 160, S)
    float* dtb   = ssm_p + (long)NHEADS * BATCH * 160 * SEQ;  // (NH, B, HD, S)
    float* yb    = dtb   + (long)NHEADS * BATCH * HEADD * SEQ;// (NH, B, HD, S)
    float* yg    = yb    + (long)NHEADS * BATCH * HEADD * SEQ;// (B, I, S)

    dim3 blk(256);

    // 1) in_proj: proj[b,o,s] = sum_h hidden[b,s,h] * in_w[o,h]   (B k-contig)
    {
        dim3 grid(SEQ / BN, ODIM / BM, BATCH);
        gemm_bf16_wmma<<<grid, blk, 0, stream>>>(
            in_w,   (long)HDIM, 0L, 0L,
            hidden, 1L, (long)HDIM, 0L, (long)SEQ * HDIM,
            proj,   (long)SEQ, 1L, 0L, (long)ODIM * SEQ,
            ODIM, SEQ, HDIM, BATCH, /*bnc=*/0, 0, nullptr, 0L);
    }
    // 2) conv + SiLU
    {
        dim3 grid(SEQ / 256, IDIM, BATCH);
        conv_silu_kernel<<<grid, blk, 0, stream>>>(proj, conv_w, conv_b, xconv);
    }
    // 3) x_proj: ssm_p[h,b,r,s] = sum_d xproj_w[h,r,d] * xconv[b,h*HD+d,s]   (B n-contig)
    {
        dim3 grid(SEQ / BN, (160 + BM - 1) / BM, NHEADS * BATCH);
        gemm_bf16_wmma<<<grid, blk, 0, stream>>>(
            xproj_w, (long)HEADD, (long)160 * HEADD, 0L,
            xconv,   (long)SEQ, 1L, (long)HEADD * SEQ, (long)IDIM * SEQ,
            ssm_p,   (long)SEQ, 1L, (long)BATCH * 160 * SEQ, (long)160 * SEQ,
            160, SEQ, HEADD, BATCH, /*bnc=*/1, 0, nullptr, 0L);
    }
    // 4) dt_proj + bias + softplus   (B n-contig)
    {
        dim3 grid(SEQ / BN, HEADD / BM, NHEADS * BATCH);
        gemm_bf16_wmma<<<grid, blk, 0, stream>>>(
            dtproj_w, (long)RRANK, (long)HEADD * RRANK, 0L,
            ssm_p,    (long)SEQ, 1L, (long)BATCH * 160 * SEQ, (long)160 * SEQ,
            dtb,      (long)SEQ, 1L, (long)BATCH * HEADD * SEQ, (long)HEADD * SEQ,
            HEADD, SEQ, RRANK, BATCH, /*bnc=*/1, 1, dtproj_b, (long)HEADD);
    }
    // 5) selective scan
    {
        dim3 grid(HEADD / 256, BATCH, NHEADS);
        scan_kernel<<<grid, blk, 0, stream>>>(ssm_p, dtb, xconv, A_log, yb);
    }
    // 6) gating
    {
        dim3 grid(SEQ / 256, IDIM, BATCH);
        gate_kernel<<<grid, blk, 0, stream>>>(yb, xconv, proj, Dp, yg);
    }
    // 7) out_proj: out[b,s,h] = sum_i out_w[h,i] * yg[b,i,s]   (B n-contig, transposed store)
    {
        dim3 grid(SEQ / BN, HDIM / BM, BATCH);
        gemm_bf16_wmma<<<grid, blk, 0, stream>>>(
            out_w, (long)IDIM, 0L, 0L,
            yg,    (long)SEQ, 1L, 0L, (long)IDIM * SEQ,
            out,   1L, (long)HDIM, 0L, (long)SEQ * HDIM,
            HDIM, SEQ, IDIM, BATCH, /*bnc=*/1, 0, nullptr, 0L);
    }
}
